// FLAT_21242908246425
// MI455X (gfx1250) — compile-verified, matrix operand order checked by reading, not run on hardware
//
#include <hip/hip_runtime.h>
#include <hip/hip_bf16.h>
#include <math.h>

// ---------------- problem constants (match reference) ----------------
constexpr int kD    = 256;
constexpr int kH    = 8;
constexpr int kDK   = 32;
constexpr int kL    = 2;
constexpr int kDFF  = 1024;
constexpr int kMAXR = 16;
constexpr int kNL   = 25;
constexpr int kB    = 16;
constexpr int kS    = 512;
constexpr int kRows = kB * kS;          // 8192 token rows
constexpr int kRelN = 2 * kMAXR + 1;    // 33
constexpr int kRelP = 48;               // padded to tile multiple

typedef __bf16 v16bf __attribute__((ext_vector_type(16)));
typedef float  v8f   __attribute__((ext_vector_type(8)));

// ---------------- WMMA fragment helpers (gfx1250 wave32 layouts) ------
// 16-bit A (16x32) / Bt (16 rows x 32 K) fragment, row-major source with
// leading dimension ld (elements). Lane m = lane&15, half = lane>>4:
//   elems 0..7  -> K = half*8 + 0..7
//   elems 8..15 -> K = 16 + half*8 + 0..7
__device__ __forceinline__ v16bf load_frag(const __bf16* __restrict__ base,
                                           int ld, int row0, int k0) {
  const int lane = threadIdx.x & 31;
  const int m    = lane & 15;
  const int hi   = lane >> 4;
  const __bf16* p = base + (size_t)(row0 + m) * (size_t)ld + k0 + hi * 8;
  v16bf f;
#pragma unroll
  for (int i = 0; i < 8; ++i) { f[i] = p[i]; f[i + 8] = p[i + 16]; }
  return f;
}

__device__ __forceinline__ v8f wmma_bf16(v16bf a, v16bf b, v8f c) {
  return __builtin_amdgcn_wmma_f32_16x16x32_bf16(false, a, false, b,
                                                 (short)0, c, false, false);
}

// ---------------- generic GEMM: C[M,N] = A[M,K] * Bt[N,K]^T -----------
// One wave per 32x(NT*16) output tile; NT is compile-time so the tile
// loop fully unrolls (no exec-mask churn, accumulators stay pinned).
// Loads of a k-step are issued before the wmma chain to clause them.
template <int NT>
__global__ __launch_bounds__(32) void wmma_gemm(
    const __bf16* __restrict__ A, const __bf16* __restrict__ Bt,
    float* __restrict__ C, __bf16* __restrict__ Cb,
    const float* __restrict__ bias,
    int K, int ldc, int n_valid, int relu) {
  const int n_base = blockIdx.x * (NT * 16);
  const int row0   = blockIdx.y * 32;
  const int lane   = threadIdx.x & 31;
  const int hi     = lane >> 4;
  const int lc     = lane & 15;

  v8f acc0[NT], acc1[NT];
#pragma unroll
  for (int t = 0; t < NT; ++t) { acc0[t] = v8f{}; acc1[t] = v8f{}; }

#pragma unroll 2
  for (int k0 = 0; k0 < K; k0 += 32) {
    v16bf a0 = load_frag(A, K, row0, k0);
    v16bf a1 = load_frag(A, K, row0 + 16, k0);
    v16bf b[NT];
#pragma unroll
    for (int t = 0; t < NT; ++t) b[t] = load_frag(Bt, K, n_base + t * 16, k0);
#pragma unroll
    for (int t = 0; t < NT; ++t) {
      acc0[t] = wmma_bf16(a0, b[t], acc0[t]);
      acc1[t] = wmma_bf16(a1, b[t], acc1[t]);
    }
  }

#pragma unroll
  for (int t = 0; t < NT; ++t) {
    const int col = n_base + t * 16 + lc;
    if (col >= n_valid) continue;
    const float bv = bias ? bias[col] : 0.0f;
#pragma unroll
    for (int r = 0; r < 8; ++r) {
      const int row_a = row0 + hi * 8 + r;
      const int row_b = row_a + 16;
      float v0 = acc0[t][r] + bv;
      float v1 = acc1[t][r] + bv;
      if (relu) { v0 = fmaxf(v0, 0.0f); v1 = fmaxf(v1, 0.0f); }
      if (C)  { C[(size_t)row_a * ldc + col] = v0;
                C[(size_t)row_b * ldc + col] = v1; }
      if (Cb) { Cb[(size_t)row_a * ldc + col] = (__bf16)v0;
                Cb[(size_t)row_b * ldc + col] = (__bf16)v1; }
    }
  }
}

// ---------------- weight prep ----------------------------------------
__global__ void transpose_w(const float* __restrict__ W, __bf16* __restrict__ Wt,
                            int K, int N) {   // W[K,N] f32 -> Wt[N,K] bf16
  int t = blockIdx.x * blockDim.x + threadIdx.x;
  if (t >= K * N) return;
  int k = t / N, n = t - k * N;
  Wt[(size_t)n * K + k] = (__bf16)W[t];
}

__global__ void conv_bf16(const float* __restrict__ s, __bf16* __restrict__ d, int n) {
  int t = blockIdx.x * blockDim.x + threadIdx.x;
  if (t < n) d[t] = (__bf16)s[t];
}

// ---------------- embedding gather -----------------------------------
__global__ void embed_gather(const int* __restrict__ cid, const int* __restrict__ bid,
                             const float* __restrict__ ce, const float* __restrict__ be,
                             __bf16* __restrict__ xcat) {
  int t = blockIdx.x * blockDim.x + threadIdx.x;     // over kRows*kD
  if (t >= kRows * kD) return;
  int d = t & (kD - 1), row = t >> 8;
  xcat[(size_t)row * (2 * kD) + d]      = (__bf16)ce[(size_t)cid[row] * kD + d];
  xcat[(size_t)row * (2 * kD) + kD + d] = (__bf16)be[(size_t)bid[row] * kD + d];
}

// ---------------- sinusoidal PE add ----------------------------------
__global__ void add_pe(float* __restrict__ x, __bf16* __restrict__ xb) {
  int t = blockIdx.x * blockDim.x + threadIdx.x;     // over kRows*kD
  if (t >= kRows * kD) return;
  int d = t & (kD - 1), row = t >> 8, s = row & (kS - 1);
  int pair = d & ~1;
  float ang = (float)s * expf((float)pair * (-9.210340371976184f / (float)kD));
  float pe  = (d & 1) ? cosf(ang) : sinf(ang);
  float v = x[t] + pe;
  x[t] = v;
  xb[t] = (__bf16)v;
}

// ---------------- head split / V transpose ---------------------------
// q,k,v: [B,S,H*DK] bf16 -> qh,kh: [B,H,S,DK], vt: [B,H,DK,S]
__global__ void headsplit(const __bf16* __restrict__ q, const __bf16* __restrict__ k,
                          const __bf16* __restrict__ v, __bf16* __restrict__ qh,
                          __bf16* __restrict__ kh, __bf16* __restrict__ vt) {
  int t = blockIdx.x * blockDim.x + threadIdx.x;     // over kRows*kD
  if (t >= kRows * kD) return;
  int d = t & 31, h = (t >> 5) & 7, s = (t >> 8) & (kS - 1), b = t >> 17;
  size_t hidx = (((size_t)(b * kH + h)) * kS + s) * kDK + d;
  size_t vidx = (((size_t)(b * kH + h)) * kDK + d) * kS + s;
  qh[hidx] = q[t];
  kh[hidx] = k[t];
  vt[vidx] = v[t];
}

// ---------------- fused attention per (b,h,i-tile) --------------------
// scores = (Q K^T)/sqrt(DK) + gather(qrel) + mask ; softmax ; ctx = P V
__global__ __launch_bounds__(128) void attn_kernel(
    const __bf16* __restrict__ Qh, const __bf16* __restrict__ Kh,
    const __bf16* __restrict__ Vt, const float* __restrict__ qrel,
    const int* __restrict__ mask, __bf16* __restrict__ ctx) {
  __shared__ float sc[16][kS];                        // 32 KB
  const int i0   = blockIdx.x * 16;
  const int h    = blockIdx.y;
  const int b    = blockIdx.z;
  const int wave = threadIdx.x >> 5;
  const int lane = threadIdx.x & 31;
  const int hi   = lane >> 4;
  const int lc   = lane & 15;
  const size_t bh = (size_t)(b * kH + h);
  const __bf16* Qbh = Qh + bh * kS * kDK;
  const __bf16* Kbh = Kh + bh * kS * kDK;
  const __bf16* Vbh = Vt + bh * kDK * kS;
  const float*  qr  = qrel + bh * kS * kRelP;
  const float scale = 0.17677669529663687f;           // 1/sqrt(32)

  // ---- scores strip [16, 512] -----------------------------------
  v16bf aq = load_frag(Qbh, kDK, i0, 0);
#pragma unroll
  for (int jt = 0; jt < 8; ++jt) {
    const int j0 = wave * 128 + jt * 16;
    v16bf bk = load_frag(Kbh, kDK, j0, 0);
    v8f acc = v8f{};
    acc = wmma_bf16(aq, bk, acc);
#pragma unroll
    for (int r = 0; r < 8; ++r) {
      const int i  = hi * 8 + r;
      const int j  = j0 + lc;
      const int ig = i0 + i;
      int dd = j - ig;
      dd = dd < -kMAXR ? -kMAXR : (dd > kMAXR ? kMAXR : dd);
      float sv = acc[r] * scale + qr[(size_t)ig * kRelP + dd + kMAXR];
      if (mask[b * kS + j] == 0) sv = -1e9f;
      sc[i][j] = sv;
    }
  }
  __syncthreads();

  // ---- softmax over j (8 lanes per row) -------------------------
  {
    const int row = threadIdx.x >> 3;
    const int g   = threadIdx.x & 7;
    float mx = -3.0e38f;
    for (int j = g; j < kS; j += 8) mx = fmaxf(mx, sc[row][j]);
#pragma unroll
    for (int o = 1; o < 8; o <<= 1) mx = fmaxf(mx, __shfl_xor(mx, o));
    float sum = 0.0f;
    for (int j = g; j < kS; j += 8) {
      float e = __expf(sc[row][j] - mx);
      sc[row][j] = e;
      sum += e;
    }
#pragma unroll
    for (int o = 1; o < 8; o <<= 1) sum += __shfl_xor(sum, o);
    const float inv = 1.0f / sum;
    for (int j = g; j < kS; j += 8) sc[row][j] *= inv;
  }
  __syncthreads();

  // ---- ctx = P @ V  (waves 0,1: one 16x16 tile each, K=512) -----
  if (wave < 2) {
    const int n0 = wave * 16;
    v8f acc = v8f{};
    for (int ks = 0; ks < kS; ks += 32) {
      v16bf a;
#pragma unroll
      for (int ii = 0; ii < 8; ++ii) {
        a[ii]     = (__bf16)sc[lc][ks + hi * 8 + ii];
        a[ii + 8] = (__bf16)sc[lc][ks + 16 + hi * 8 + ii];
      }
      v16bf bv = load_frag(Vbh, kS, n0, ks);
      acc = wmma_bf16(a, bv, acc);
    }
#pragma unroll
    for (int r = 0; r < 8; ++r) {
      const int i = i0 + hi * 8 + r;
      const int d = n0 + lc;
      ctx[((size_t)(b * kS + i)) * kD + h * kDK + d] = (__bf16)acc[r];
    }
  }
}

// ---------------- residual + LayerNorm (wave per row) -----------------
__global__ __launch_bounds__(32) void add_ln(
    const float* __restrict__ a, const float* __restrict__ res,
    const float* __restrict__ g, const float* __restrict__ be,
    float* __restrict__ of, __bf16* __restrict__ ob) {
  const int row  = blockIdx.x;
  const int lane = threadIdx.x;
  const float* pa = a + (size_t)row * kD;
  const float* pr = res + (size_t)row * kD;
  float vals[8];
  float sum = 0.0f;
#pragma unroll
  for (int i = 0; i < 8; ++i) {
    float v = pa[lane + i * 32] + pr[lane + i * 32];
    vals[i] = v;
    sum += v;
  }
#pragma unroll
  for (int o = 1; o < 32; o <<= 1) sum += __shfl_xor(sum, o);
  const float mean = sum * (1.0f / kD);
  float var = 0.0f;
#pragma unroll
  for (int i = 0; i < 8; ++i) { float d = vals[i] - mean; var += d * d; }
#pragma unroll
  for (int o = 1; o < 32; o <<= 1) var += __shfl_xor(var, o);
  const float rs = rsqrtf(var * (1.0f / kD) + 1e-5f);
#pragma unroll
  for (int i = 0; i < 8; ++i) {
    const int c = lane + i * 32;
    float y = (vals[i] - mean) * rs * g[c] + be[c];
    of[(size_t)row * kD + c] = y;
    ob[(size_t)row * kD + c] = (__bf16)y;
  }
}

// ---------------- host orchestration ----------------------------------
extern "C" void kernel_launch(void* const* d_in, const int* in_sizes, int n_in,
                              void* d_out, int out_size, void* d_ws, size_t ws_size,
                              hipStream_t stream) {
  (void)in_sizes; (void)n_in; (void)out_size; (void)ws_size;
  // inputs (setup_inputs order)
  const int*   char_ids   = (const int*)  d_in[0];
  const int*   bigram_ids = (const int*)  d_in[1];
  const int*   mask       = (const int*)  d_in[2];
  const float* char_emb   = (const float*)d_in[3];
  const float* bigram_emb = (const float*)d_in[4];
  const float* comb_W     = (const float*)d_in[5];
  const float* comb_b     = (const float*)d_in[6];
  const float* Wq         = (const float*)d_in[7];
  const float* bq         = (const float*)d_in[8];
  const float* Wk         = (const float*)d_in[9];
  const float* bk         = (const float*)d_in[10];
  const float* Wv         = (const float*)d_in[11];
  const float* bv         = (const float*)d_in[12];
  const float* Wo         = (const float*)d_in[13];
  const float* bo         = (const float*)d_in[14];
  const float* rel_emb    = (const float*)d_in[15];
  const float* ln1_g      = (const float*)d_in[16];
  const float* ln1_b      = (const float*)d_in[17];
  const float* W1         = (const float*)d_in[18];
  const float* b1         = (const float*)d_in[19];
  const float* W2         = (const float*)d_in[20];
  const float* b2         = (const float*)d_in[21];
  const float* ln2_g      = (const float*)d_in[22];
  const float* ln2_b      = (const float*)d_in[23];
  const float* cls_W      = (const float*)d_in[24];
  const float* cls_b      = (const float*)d_in[25];

  // ---- workspace bump allocator ----
  char* base = (char*)d_ws;
  size_t off = 0;
  auto alloc = [&](size_t bytes) -> char* {
    off = (off + 255) & ~(size_t)255;
    char* p = base + off;
    off += bytes;
    return p;
  };
  __bf16* xcat    = (__bf16*)alloc((size_t)kRows * 2 * kD * 2);
  float*  x_f32   = (float*) alloc((size_t)kRows * kD * 4);
  __bf16* x_bf16  = (__bf16*)alloc((size_t)kRows * kD * 2);
  __bf16* combWt  = (__bf16*)alloc((size_t)kD * 2 * kD * 2);
  __bf16* wqt     = (__bf16*)alloc((size_t)kL * kD * kD * 2);
  __bf16* wkt     = (__bf16*)alloc((size_t)kL * kD * kD * 2);
  __bf16* wvt     = (__bf16*)alloc((size_t)kL * kD * kD * 2);
  __bf16* wot     = (__bf16*)alloc((size_t)kL * kD * kD * 2);
  __bf16* w1t     = (__bf16*)alloc((size_t)kL * kDFF * kD * 2);
  __bf16* w2t     = (__bf16*)alloc((size_t)kL * kD * kDFF * 2);
  __bf16* relt    = (__bf16*)alloc((size_t)kL * kRelP * kDK * 2);
  __bf16* clsWt   = (__bf16*)alloc((size_t)32 * kD * 2);
  __bf16* qb      = (__bf16*)alloc((size_t)kRows * kD * 2);
  __bf16* kb      = (__bf16*)alloc((size_t)kRows * kD * 2);
  __bf16* vb      = (__bf16*)alloc((size_t)kRows * kD * 2);
  __bf16* qh      = (__bf16*)alloc((size_t)kRows * kD * 2);
  __bf16* kh      = (__bf16*)alloc((size_t)kRows * kD * 2);
  __bf16* vt      = (__bf16*)alloc((size_t)kRows * kD * 2);
  float*  qrel    = (float*) alloc((size_t)kB * kH * kS * kRelP * 4);
  __bf16* ctx     = (__bf16*)alloc((size_t)kRows * kD * 2);
  float*  tmp_f32 = (float*) alloc((size_t)kRows * kD * 4);
  __bf16* ffh     = (__bf16*)alloc((size_t)kRows * kDFF * 2);

  auto cdiv = [](int a, int b) { return (a + b - 1) / b; };

  // ---- weight prep (bf16, [N,K] layouts) ----
  {
    int n = 2 * kD * kD;
    transpose_w<<<cdiv(n, 256), 256, 0, stream>>>(comb_W, combWt, 2 * kD, kD);
    for (int l = 0; l < kL; ++l) {
      int nd = kD * kD;
      transpose_w<<<cdiv(nd, 256), 256, 0, stream>>>(Wq + (size_t)l * nd, wqt + (size_t)l * nd, kD, kD);
      transpose_w<<<cdiv(nd, 256), 256, 0, stream>>>(Wk + (size_t)l * nd, wkt + (size_t)l * nd, kD, kD);
      transpose_w<<<cdiv(nd, 256), 256, 0, stream>>>(Wv + (size_t)l * nd, wvt + (size_t)l * nd, kD, kD);
      transpose_w<<<cdiv(nd, 256), 256, 0, stream>>>(Wo + (size_t)l * nd, wot + (size_t)l * nd, kD, kD);
      int nf = kD * kDFF;
      transpose_w<<<cdiv(nf, 256), 256, 0, stream>>>(W1 + (size_t)l * nf, w1t + (size_t)l * nf, kD, kDFF);
      transpose_w<<<cdiv(nf, 256), 256, 0, stream>>>(W2 + (size_t)l * nf, w2t + (size_t)l * nf, kDFF, kD);
      int nr = kRelN * kDK;   // rel_emb is already [dist, DK] == Bt layout
      conv_bf16<<<cdiv(nr, 256), 256, 0, stream>>>(rel_emb + (size_t)l * nr,
                                                   relt + (size_t)l * kRelP * kDK, nr);
    }
    int nc = kD * kNL;
    transpose_w<<<cdiv(nc, 256), 256, 0, stream>>>(cls_W, clsWt, kD, kNL);
  }

  // ---- embeddings + combine + positional encoding ----
  embed_gather<<<cdiv(kRows * kD, 256), 256, 0, stream>>>(char_ids, bigram_ids,
                                                          char_emb, bigram_emb, xcat);
  wmma_gemm<4><<<dim3(kD / 64, kRows / 32), 32, 0, stream>>>(
      xcat, combWt, x_f32, nullptr, comb_b, 2 * kD, kD, kD, 0);
  add_pe<<<cdiv(kRows * kD, 256), 256, 0, stream>>>(x_f32, x_bf16);

  // ---- transformer layers ----
  for (int l = 0; l < kL; ++l) {
    const size_t wdd = (size_t)l * kD * kD;
    const size_t wdf = (size_t)l * kD * kDFF;
    // Q, K, V projections (bf16 outputs, [B,S,H*DK])
    wmma_gemm<4><<<dim3(kD / 64, kRows / 32), 32, 0, stream>>>(
        x_bf16, wqt + wdd, nullptr, qb, bq + (size_t)l * kD, kD, kD, kD, 0);
    wmma_gemm<4><<<dim3(kD / 64, kRows / 32), 32, 0, stream>>>(
        x_bf16, wkt + wdd, nullptr, kb, bk + (size_t)l * kD, kD, kD, kD, 0);
    wmma_gemm<4><<<dim3(kD / 64, kRows / 32), 32, 0, stream>>>(
        x_bf16, wvt + wdd, nullptr, vb, bv + (size_t)l * kD, kD, kD, kD, 0);
    headsplit<<<cdiv(kRows * kD, 256), 256, 0, stream>>>(qb, kb, vb, qh, kh, vt);
    // qrel[B*H*S, 33(pad 48)] = Qh @ rel_emb^T   (NT=3 -> N covered exactly)
    wmma_gemm<3><<<dim3(1, (kB * kH * kS) / 32), 32, 0, stream>>>(
        qh, relt + (size_t)l * kRelP * kDK, qrel, nullptr, nullptr,
        kDK, kRelP, kRelP, 0);
    // fused attention
    attn_kernel<<<dim3(kS / 16, kH, kB), 128, 0, stream>>>(qh, kh, vt, qrel, mask, ctx);
    // output projection + residual + LN1
    wmma_gemm<4><<<dim3(kD / 64, kRows / 32), 32, 0, stream>>>(
        ctx, wot + wdd, tmp_f32, nullptr, bo + (size_t)l * kD, kD, kD, kD, 0);
    add_ln<<<kRows, 32, 0, stream>>>(tmp_f32, x_f32,
                                     ln1_g + (size_t)l * kD, ln1_b + (size_t)l * kD,
                                     x_f32, x_bf16);
    // FFN
    wmma_gemm<4><<<dim3(kDFF / 64, kRows / 32), 32, 0, stream>>>(
        x_bf16, w1t + wdf, nullptr, ffh, b1 + (size_t)l * kDFF, kD, kDFF, kDFF, 1);
    wmma_gemm<4><<<dim3(kD / 64, kRows / 32), 32, 0, stream>>>(
        ffh, w2t + wdf, tmp_f32, nullptr, b2 + (size_t)l * kD, kDFF, kD, kD, 0);
    add_ln<<<kRows, 32, 0, stream>>>(tmp_f32, x_f32,
                                     ln2_g + (size_t)l * kD, ln2_b + (size_t)l * kD,
                                     x_f32, x_bf16);
  }

  // ---- classifier head (N padded 25 -> 32, stores guarded) ----
  wmma_gemm<2><<<dim3(1, kRows / 32), 32, 0, stream>>>(
      x_bf16, clsWt, (float*)d_out, nullptr, cls_b, kD, kNL, kNL, 0);
}
